// Bi_concat_AttentionFusion_51256139710860
// MI455X (gfx1250) — matrix operand order repaired and checked by reading, not compiled, not run on hardware
//
#include <hip/hip_runtime.h>

// ---------------------------------------------------------------------------
// Bi-directional cross attention, fused flash-attention bf16 WMMA pipeline
// for gfx1250 (MI455X).
// out = 0.5*( softmax(G L^T) L  +  softmax(L G^T) G ),  B=8, N=2048, D=768.
//
// Pipeline:
//   cvt_rm : f32 -> bf16 row-major copies of G, L      (A-frags, K B-frags)
//   cvt_tr : f32 -> bf16 column-major copies of G, L   (V B-frags, contiguous)
//   flash  : fused online-softmax attention, one launch per direction.
// ---------------------------------------------------------------------------

typedef unsigned short u16;
typedef __attribute__((ext_vector_type(16))) __bf16 v16bf;
typedef __attribute__((ext_vector_type(8)))  float  v8f;

namespace {

constexpr int Bb = 8;
constexpr int N  = 2048;
constexpr int D  = 768;

constexpr int BQ = 32;   // query rows per workgroup
constexpr int KT = 64;   // keys per main-loop iteration
constexpr int KC = 32;   // WMMA K per instruction (bf16)
constexpr int NKC = D / KC;     // 24 k-chunks across D

// LDS pitches (u16 elements); row strides are multiples of 8 u16 = 16B
constexpr int QP = D + 8;       // 776  -> 1552B
constexpr int PP = KT + 8;      // 72   -> 144B

constexpr size_t SMEM_FLASH =
    ((size_t)BQ * QP + (size_t)BQ * PP) * 2 + (size_t)4 * BQ * 2 * 4;

union FragU { v16bf v; uint4 q[2]; };

__device__ __forceinline__ u16 f2bf(float x) {
  unsigned u = __float_as_uint(x);
  unsigned r = u + 0x7FFFu + ((u >> 16) & 1u);   // round-to-nearest-even
  return (u16)(r >> 16);
}

// A-matrix fragment (16x32 bf16): lane m=l%16 is row M, half h=l/16 selects
// K 0..15 / 16..31 (contiguous).
// p must point at  base + m*ld + h*16.
__device__ __forceinline__ v16bf ld_fragA(const u16* p) {
  FragU f;
  f.q[0] = *reinterpret_cast<const uint4*>(p);
  f.q[1] = *reinterpret_cast<const uint4*>(p + 8);
  return f.v;
}

// B-matrix fragment (32x16 bf16): lane n=l%16 is column N, half h=l/16 selects
// K 0..15 / 16..31 (contiguous).  p must point at base + n*ld + h*16.
__device__ __forceinline__ v16bf ld_fragB(const u16* p) {
  FragU f;
  f.q[0] = *reinterpret_cast<const uint4*>(p);
  f.q[1] = *reinterpret_cast<const uint4*>(p + 8);
  return f.v;
}

__device__ __forceinline__ v8f wmma_bf16(v16bf a, v16bf b, v8f c) {
  return __builtin_amdgcn_wmma_f32_16x16x32_bf16(
      false, a, false, b, (short)0, c, false, false);
}

// ---------------------------------------------------------------------------
// Kernel 0: f32 -> bf16 row-major (vectorized x4)
// ---------------------------------------------------------------------------
__global__ __launch_bounds__(256) void cvt_rm_kernel(
    const float* __restrict__ src, u16* __restrict__ dst, int n4) {
  int i = blockIdx.x * 256 + threadIdx.x;
  if (i < n4) {
    float4 f = reinterpret_cast<const float4*>(src)[i];
    ushort4 o;
    o.x = f2bf(f.x); o.y = f2bf(f.y); o.z = f2bf(f.z); o.w = f2bf(f.w);
    reinterpret_cast<ushort4*>(dst)[i] = o;
  }
}

// ---------------------------------------------------------------------------
// Kernel 1: f32 [B][N][D] -> bf16 [B][D][N] tiled transpose (64x64 via LDS)
// ---------------------------------------------------------------------------
__global__ __launch_bounds__(256) void cvt_tr_kernel(
    const float* __restrict__ src, u16* __restrict__ dst) {
  __shared__ u16 tile[64][72];                  // 72 u16 pitch = 144B
  const int tid = threadIdx.x;
  const int b  = blockIdx.z;
  const int r0 = blockIdx.x * 64;               // row (N) tile origin
  const int c0 = blockIdx.y * 64;               // col (D) tile origin

  // Coalesced read (float4) + bf16 convert into LDS tile
  for (int i = tid; i < 64 * 16; i += 256) {    // 64 rows x 16 float4
    int row = i >> 4, cv = (i & 15) * 4;
    float4 f = *reinterpret_cast<const float4*>(
        src + ((size_t)b * N + r0 + row) * D + c0 + cv);
    ushort4 o;
    o.x = f2bf(f.x); o.y = f2bf(f.y); o.z = f2bf(f.z); o.w = f2bf(f.w);
    *reinterpret_cast<ushort4*>(&tile[row][cv]) = o;
  }
  __syncthreads();

  // Coalesced transposed write: dst[(b*D + d)*N + row], 16B per store
  for (int i = tid; i < 64 * 8; i += 256) {     // 64 cols x 8 groups of 8 rows
    int col = i >> 3, rv = (i & 7) * 8;
    union { uint4 v; u16 e[8]; } o;
#pragma unroll
    for (int j = 0; j < 8; ++j) o.e[j] = tile[rv + j][col];
    *reinterpret_cast<uint4*>(dst + ((size_t)b * D + c0 + col) * N + r0 + rv) = o.v;
  }
}

// ---------------------------------------------------------------------------
// Kernel 2: fused flash attention.   out (+)= 0.5 * softmax(Q K^T) V
// grid (N/BQ, B), block 256 (8 waves).  wave = (qsub in 0..1, sub4 in 0..3)
//   Phase A: (qsub, ksub=sub4) computes 16x16 S subtile over keys
//            [kt+16*ksub, +16), block-wide online softmax via LDS TM/TS,
//            writes unnormalized P (bf16) to LDS.
//   Phase B: (qsub, dsub=sub4) rescales acc and accumulates P * V over a
//            192-wide d slice, V B-frags streamed from the column-major copy.
// ---------------------------------------------------------------------------
template <bool ACC>
__global__ __launch_bounds__(256) void flash_kernel(
    const u16* __restrict__ Qrm, const u16* __restrict__ Krm,
    const u16* __restrict__ Vcm, float* __restrict__ out) {
  extern __shared__ u16 smem[];
  u16*   Qs = smem;                                        // [BQ][QP]
  u16*   Ps = smem + (size_t)BQ * QP;                      // [BQ][PP]
  float* TM = reinterpret_cast<float*>(Ps + (size_t)BQ * PP);  // [4][BQ]
  float* TS = TM + 4 * BQ;                                     // [4][BQ]

  const int tid  = threadIdx.x;
  const int w    = tid >> 5, lane = tid & 31;
  const int qsub = w >> 2,   sub4 = w & 3;
  const int m15  = lane & 15, h = lane >> 4;
  const int q0   = blockIdx.x * BQ;
  const int b    = blockIdx.y;

  // Stage Q tile (coalesced 16B copies)
  for (int i = tid; i < BQ * (D / 8); i += 256) {
    int row = i / (D / 8), dv = (i % (D / 8)) * 8;
    *reinterpret_cast<uint4*>(Qs + row * QP + dv) =
        *reinterpret_cast<const uint4*>(Qrm + ((size_t)b * N + q0 + row) * D + dv);
  }

  float m_run[8], s_run[8];
#pragma unroll
  for (int r = 0; r < 8; ++r) { m_run[r] = -3.0e38f; s_run[r] = 0.f; }

  v8f acc[12];
#pragma unroll
  for (int j = 0; j < 12; ++j) acc[j] = (v8f){0.f, 0.f, 0.f, 0.f, 0.f, 0.f, 0.f, 0.f};

  const u16* Qbase = Qs + (qsub * 16 + m15) * QP + h * 16;
  __syncthreads();

  for (int kt = 0; kt < N; kt += KT) {
    // ---- Phase A: S subtile ----
    const u16* Kbase = Krm + ((size_t)b * N + kt + sub4 * 16 + m15) * D + h * 16;
    v8f c = {0.f, 0.f, 0.f, 0.f, 0.f, 0.f, 0.f, 0.f};
#pragma unroll
    for (int kc = 0; kc < NKC; ++kc)
      c = wmma_bf16(ld_fragA(Qbase + kc * KC), ld_fragB(Kbase + kc * KC), c);

    // per-row max of this 16x16 subtile -> LDS
#pragma unroll
    for (int r = 0; r < 8; ++r) {
      float t = c[r];
      t = fmaxf(t, __shfl_xor(t, 8, 16));
      t = fmaxf(t, __shfl_xor(t, 4, 16));
      t = fmaxf(t, __shfl_xor(t, 2, 16));
      t = fmaxf(t, __shfl_xor(t, 1, 16));
      if (m15 == 0) TM[sub4 * BQ + qsub * 16 + r + 8 * h] = t;
    }
    __syncthreads();

    // block-wide new running max, unnormalized P, per-subtile sums
    float mnew[8], scale[8];
#pragma unroll
    for (int r = 0; r < 8; ++r) {
      int row = qsub * 16 + r + 8 * h;
      float tm = fmaxf(fmaxf(TM[row], TM[BQ + row]),
                       fmaxf(TM[2 * BQ + row], TM[3 * BQ + row]));
      float mn = fmaxf(m_run[r], tm);
      mnew[r]  = mn;
      scale[r] = __expf(m_run[r] - mn);
      m_run[r] = mn;
    }
#pragma unroll
    for (int r = 0; r < 8; ++r) {
      int row = qsub * 16 + r + 8 * h;
      float e = __expf(c[r] - mnew[r]);
      Ps[row * PP + sub4 * 16 + m15] = f2bf(e);
      e += __shfl_xor(e, 8, 16);
      e += __shfl_xor(e, 4, 16);
      e += __shfl_xor(e, 2, 16);
      e += __shfl_xor(e, 1, 16);
      if (m15 == 0) TS[sub4 * BQ + row] = e;
    }
    __syncthreads();

    // update running sums; rescale accumulators
#pragma unroll
    for (int r = 0; r < 8; ++r) {
      int row = qsub * 16 + r + 8 * h;
      float tsum = (TS[row] + TS[BQ + row]) + (TS[2 * BQ + row] + TS[3 * BQ + row]);
      s_run[r] = s_run[r] * scale[r] + tsum;
    }
#pragma unroll
    for (int j = 0; j < 12; ++j)
#pragma unroll
      for (int r = 0; r < 8; ++r) acc[j][r] *= scale[r];

    // ---- Phase B: acc += P(16xKT) * V(KT x 192-slice) ----
#pragma unroll
    for (int kc2 = 0; kc2 < KT / KC; ++kc2) {
      v16bf a = ld_fragA(Ps + (qsub * 16 + m15) * PP + kc2 * KC + h * 16);
#pragma unroll
      for (int j = 0; j < 12; ++j) {
        v16bf bb = ld_fragB(Vcm + ((size_t)b * D + sub4 * 192 + j * 16 + m15) * N
                            + kt + kc2 * KC + h * 16);
        acc[j] = wmma_bf16(a, bb, acc[j]);
      }
    }
    // (no extra barrier needed: next TM write races with nothing, and Ps/TS
    //  writes of the next iteration happen only after the next sync-a, which
    //  all waves reach only after finishing this phase B.)
  }

  // ---- Epilogue: out (+)= 0.5 * acc / s ----
  float inv[8];
#pragma unroll
  for (int r = 0; r < 8; ++r) inv[r] = 0.5f / s_run[r];
#pragma unroll
  for (int j = 0; j < 12; ++j) {
#pragma unroll
    for (int r = 0; r < 8; ++r) {
      int row = qsub * 16 + r + 8 * h;
      int col = sub4 * 192 + j * 16 + m15;
      size_t o = ((size_t)b * N + q0 + row) * (size_t)D + col;
      float v = acc[j][r] * inv[r];
      out[o] = ACC ? (out[o] + v) : v;
    }
  }
}

}  // namespace

// ---------------------------------------------------------------------------
extern "C" void kernel_launch(void* const* d_in, const int* in_sizes, int n_in,
                              void* d_out, int out_size, void* d_ws, size_t ws_size,
                              hipStream_t stream) {
  (void)in_sizes; (void)n_in; (void)out_size; (void)ws_size;
  const float* G = (const float*)d_in[0];
  const float* L = (const float*)d_in[1];
  float* out = (float*)d_out;

  const size_t T = (size_t)Bb * N * D;   // elements per tensor
  u16* Grm = (u16*)d_ws;                 // [B][N][D]
  u16* Lrm = Grm + T;
  u16* Gcm = Lrm + T;                    // [B][D][N]
  u16* Lcm = Gcm + T;

  const int n4 = (int)(T / 4);
  cvt_rm_kernel<<<n4 / 256, 256, 0, stream>>>(G, Grm, n4);
  cvt_rm_kernel<<<n4 / 256, 256, 0, stream>>>(L, Lrm, n4);

  dim3 tgrid(N / 64, D / 64, Bb);
  cvt_tr_kernel<<<tgrid, 256, 0, stream>>>(G, Gcm);
  cvt_tr_kernel<<<tgrid, 256, 0, stream>>>(L, Lcm);

  dim3 grid(N / BQ, Bb);
  // dir 0: Q=G, K=V=L (attended_local); dir 1: Q=L, K=V=G (attended_global)
  flash_kernel<false><<<grid, 256, SMEM_FLASH, stream>>>(Grm, Lrm, Lcm, out);
  flash_kernel<true ><<<grid, 256, SMEM_FLASH, stream>>>(Lrm, Grm, Gcm, out);
}
